// self_attention_43465069035651
// MI455X (gfx1250) — matrix operand order repaired and checked by reading, not compiled, not run on hardware
//
#include <hip/hip_runtime.h>
#include <hip/hip_bf16.h>

typedef __attribute__((ext_vector_type(16))) _Float16 v16h;
typedef __attribute__((ext_vector_type(8)))  _Float16 v8h;
typedef __attribute__((ext_vector_type(8)))  float    v8f;
typedef __attribute__((ext_vector_type(4)))  int      v4i;

#define EMBED 1024
#define HEADS 16
#define HDIM  64
#define BATCH 2
#define SEQ   2048
#define NROWS (BATCH*SEQ*HEADS)   // 65536 (b,s,h) rows

#if __has_builtin(__builtin_amdgcn_global_load_async_to_lds_b128) && \
    __has_builtin(__builtin_amdgcn_s_wait_asynccnt)
#define HAVE_ASYNC_LDS 1
#else
#define HAVE_ASYNC_LDS 0
#endif

// 16-byte global -> LDS tile copy (async on CDNA5 if toolchain exposes it)
__device__ __forceinline__ void copy16_g2l(const _Float16* g, _Float16* l) {
#if HAVE_ASYNC_LDS
  __builtin_amdgcn_global_load_async_to_lds_b128(
      (__attribute__((address_space(1))) v4i*)g,
      (__attribute__((address_space(3))) v4i*)l, 0, 0);
#else
  *(v8h*)l = *(const v8h*)g;
#endif
}
__device__ __forceinline__ void tile_join() {
#if HAVE_ASYNC_LDS
  __builtin_amdgcn_s_wait_asynccnt(0);
#endif
  __syncthreads();
}

__device__ __forceinline__ v8f wmma16(v16h a, v16h b, v8f c) {
  // D = A(16x32 f16) * B(32x16 f16) + C(16x16 f32)
  return __builtin_amdgcn_wmma_f32_16x16x32_f16(false, a, false, b, (short)0, c,
                                                false, false);
}

// A matrix 16x32 (MxK), row-major with stride ld (halves, multiple of 8).
// Lane layout (ISA 7.12.2): lanes 0-15 row=lane, halves K 0..7 then 16..23;
// lanes 16-31 same rows, K +8 (8..15, 24..31).  Two contiguous 16B runs.
__device__ __forceinline__ v16h load_a_frag(const _Float16* base, int ld) {
  const int lane = threadIdx.x & 31;
  const _Float16* rp = base + (lane & 15) * ld + ((lane & 16) ? 8 : 0);
  v8h lo = *(const v8h*)(rp);        // K 0..7   (+kofs)
  v8h hi = *(const v8h*)(rp + 16);   // K 16..23 (+kofs)
  return __builtin_shufflevector(lo, hi, 0,1,2,3,4,5,6,7,8,9,10,11,12,13,14,15);
}

// B matrix 32x16 (KxN) with element(k,n) = base[n*ld + k] (i.e. stored N-major).
// Lane layout: column N=lane&15; lanes 0-15 K 0..7 then 16..23, lanes 16-31 K 8..15 then 24..31.
// Two contiguous 16B runs per lane.
__device__ __forceinline__ v16h load_bT_frag(const _Float16* base, int ld) {
  const int lane = threadIdx.x & 31;
  const _Float16* rp = base + (lane & 15) * ld + ((lane & 16) ? 8 : 0);
  v8h lo = *(const v8h*)(rp);
  v8h hi = *(const v8h*)(rp + 16);
  return __builtin_shufflevector(lo, hi, 0,1,2,3,4,5,6,7,8,9,10,11,12,13,14,15);
}

// ---------------------------------------------------------------------------
// Wo[1024][1024] f32 -> WoT[1024][1024] f16 (transposed), 32x32 LDS tiles.
// ---------------------------------------------------------------------------
__global__ __launch_bounds__(256) void transpose_cvt(const float* __restrict__ src,
                                                     _Float16* __restrict__ dst) {
  __shared__ float t[32][33];
  const int bx = blockIdx.x, by = blockIdx.y;
  const int tx = threadIdx.x & 31, ty = threadIdx.x >> 5;   // 32 x 8
#pragma unroll
  for (int i = 0; i < 4; ++i)
    t[ty + i*8][tx] = src[(size_t)(by*32 + ty + i*8)*EMBED + bx*32 + tx];
  __syncthreads();
#pragma unroll
  for (int i = 0; i < 4; ++i)
    dst[(size_t)(bx*32 + ty + i*8)*EMBED + by*32 + tx] = (_Float16)t[tx][ty + i*8];
}

// ---------------------------------------------------------------------------
// Kernel 1: QKV projection.  rows = flat (b,s,h) so x is row-major [65536][64].
// Weights held N-major (transposed) in LDS so B-fragments are contiguous.
// Q/K/V written [B,H,S,64] f16 for contiguous streaming in attention.
// ---------------------------------------------------------------------------
__global__ __launch_bounds__(256) void qkv_kernel(
    const float* __restrict__ x,
    const float* __restrict__ Wq, const float* __restrict__ Wk,
    const float* __restrict__ Wv,
    _Float16* __restrict__ Qd, _Float16* __restrict__ Kd,
    _Float16* __restrict__ Vd) {
  __shared__ __align__(16) _Float16 xs[128][72];
  __shared__ __align__(16) _Float16 wsmT[3][64][72];   // [m][e][d]
  const int tid = threadIdx.x;
  const int blk = blockIdx.x;

  const float4* xsrc = (const float4*)(x + (size_t)blk * 128 * 64);
  for (int j = tid; j < 1024; j += 256) {        // 8 f32 -> v8h per step
    float4 lo4 = xsrc[j*2], hi4 = xsrc[j*2 + 1];
    int row = j >> 3, c = (j & 7) << 3;
    v8h h;
    h[0]=(_Float16)lo4.x; h[1]=(_Float16)lo4.y; h[2]=(_Float16)lo4.z; h[3]=(_Float16)lo4.w;
    h[4]=(_Float16)hi4.x; h[5]=(_Float16)hi4.y; h[6]=(_Float16)hi4.z; h[7]=(_Float16)hi4.w;
    *(v8h*)&xs[row][c] = h;
  }
  for (int j = tid; j < 4096; j += 256) {        // transpose: W[d][e] -> [e][d]
    int d = j >> 6, e = j & 63;
    wsmT[0][e][d] = (_Float16)Wq[j];
    wsmT[1][e][d] = (_Float16)Wk[j];
    wsmT[2][e][d] = (_Float16)Wv[j];
  }
  __syncthreads();

  const int wave = tid >> 5, lane = tid & 31;
  v16h a0 = load_a_frag(&xs[wave*16][0],  72);
  v16h a1 = load_a_frag(&xs[wave*16][32], 72);

  for (int m = 0; m < 3; ++m) {
    _Float16* __restrict__ Od = (m == 0) ? Qd : (m == 1) ? Kd : Vd;
#pragma unroll
    for (int t = 0; t < 4; ++t) {
      v8f acc = {};
      acc = wmma16(a0, load_bT_frag(&wsmT[m][t*16][0],  72), acc);
      acc = wmma16(a1, load_bT_frag(&wsmT[m][t*16][32], 72), acc);
#pragma unroll
      for (int r = 0; r < 8; ++r) {
        int mrow = r + ((lane & 16) ? 8 : 0);
        int g = blk*128 + wave*16 + mrow;        // flat (b,s,h)
        int h = g & 15, s = (g >> 4) & (SEQ-1), b = g >> 15;
        Od[((((size_t)b*HEADS + h)*SEQ + s) << 6) + t*16 + (lane & 15)] =
            (_Float16)acc[r];
      }
    }
  }
}

// ---------------------------------------------------------------------------
// Kernel 2: flash attention. grid (S/128, B*H), 8 waves x 16 query rows.
// K tile row-major (QK^T B-frag is contiguous); V tile stored d-major in LDS
// so the P*V B-frag is contiguous too.  Logit scale = 1/sqrt(EMBED) = 1/32.
// ---------------------------------------------------------------------------
__global__ __launch_bounds__(256) void attn_kernel(
    const _Float16* __restrict__ Qd, const _Float16* __restrict__ Kd,
    const _Float16* __restrict__ Vd, _Float16* __restrict__ Od) {
  __shared__ __align__(16) _Float16 ks[32][72];      // [key][d]
  __shared__ __align__(16) _Float16 vsT[64][40];     // [d][key]
  __shared__ __align__(16) _Float16 ps[8][16][40];   // per-wave P staging
  const int tid = threadIdx.x, wave = tid >> 5, lane = tid & 31;
  const int qt = blockIdx.x, bh = blockIdx.y;
  const size_t head_base = (size_t)bh * SEQ * HDIM;
  const _Float16* Kh = Kd + head_base;
  const _Float16* Vh = Vd + head_base;

  const _Float16* qrow = Qd + head_base + (size_t)(qt*128 + wave*16) * HDIM;
  v16h q0 = load_a_frag(qrow,      HDIM);
  v16h q1 = load_a_frag(qrow + 32, HDIM);

  v8f acc[4] = {};
  float mrun[8], lrun[8];
#pragma unroll
  for (int r = 0; r < 8; ++r) { mrun[r] = -3.0e38f; lrun[r] = 0.0f; }
  const float scale = 0.03125f;                      // 1/sqrt(1024)

  const int krow = tid >> 3, kc = (tid & 7) << 3;    // 256 thr x 8 halves = 32x64

  for (int kb = 0; kb < SEQ/32; ++kb) {
    const _Float16* kt = Kh + (size_t)kb * 32 * HDIM;
    const _Float16* vt = Vh + (size_t)kb * 32 * HDIM;
    copy16_g2l(kt + tid*8, &ks[krow][kc]);           // async K -> LDS
    {
      v8h hv = *(const v8h*)(vt + tid*8);            // V row chunk, transpose in
#pragma unroll
      for (int i = 0; i < 8; ++i) vsT[kc + i][krow] = hv[i];
    }
    tile_join();

    v8f s0 = {}, s1 = {};
    s0 = wmma16(q0, load_bT_frag(&ks[0][0],   72), s0);   // keys 0-15
    s0 = wmma16(q1, load_bT_frag(&ks[0][32],  72), s0);
    s1 = wmma16(q0, load_bT_frag(&ks[16][0],  72), s1);   // keys 16-31
    s1 = wmma16(q1, load_bT_frag(&ks[16][32], 72), s1);

    float alpha[8];
#pragma unroll
    for (int r = 0; r < 8; ++r) {
      float e0 = s0[r] * scale, e1 = s1[r] * scale;
      float mx = fmaxf(e0, e1);                      // row lives on 16 lanes
      mx = fmaxf(mx, __shfl_xor(mx, 1, 32));
      mx = fmaxf(mx, __shfl_xor(mx, 2, 32));
      mx = fmaxf(mx, __shfl_xor(mx, 4, 32));
      mx = fmaxf(mx, __shfl_xor(mx, 8, 32));
      float mn = fmaxf(mrun[r], mx);
      float al = __expf(mrun[r] - mn);
      float p0 = __expf(e0 - mn), p1 = __expf(e1 - mn);
      float rs = p0 + p1;
      rs += __shfl_xor(rs, 1, 32);
      rs += __shfl_xor(rs, 2, 32);
      rs += __shfl_xor(rs, 4, 32);
      rs += __shfl_xor(rs, 8, 32);
      lrun[r] = lrun[r]*al + rs;
      mrun[r] = mn;
      alpha[r] = al;
      int prow = r + ((lane & 16) ? 8 : 0), pc = lane & 15;
      ps[wave][prow][pc]      = (_Float16)p0;        // C-layout -> LDS
      ps[wave][prow][pc + 16] = (_Float16)p1;
    }
    __syncthreads();

    v16h pf = load_a_frag(&ps[wave][0][0], 40);      // LDS -> A-fragment layout
#pragma unroll
    for (int t = 0; t < 4; ++t) {
#pragma unroll
      for (int r = 0; r < 8; ++r) acc[t][r] *= alpha[r];
      acc[t] = wmma16(pf, load_bT_frag(&vsT[t*16][0], 40), acc[t]);
    }
    __syncthreads();
  }

  const int b = bh >> 4, h = bh & 15;
  const int qbase = qt*128 + wave*16;
  float inv[8];
#pragma unroll
  for (int r = 0; r < 8; ++r) inv[r] = 1.0f / lrun[r];
#pragma unroll
  for (int t = 0; t < 4; ++t) {
#pragma unroll
    for (int r = 0; r < 8; ++r) {
      int mrow = r + ((lane & 16) ? 8 : 0);
      Od[((size_t)(b*SEQ + qbase + mrow))*EMBED + h*HDIM + t*16 + (lane & 15)] =
          (_Float16)(acc[t][r] * inv[r]);
    }
  }
}

// ---------------------------------------------------------------------------
// Kernel 3: out = attn_out[4096,1024] @ Wo + bo.  Wo is pre-transposed f16 in
// global, so both LDS tiles fill with contiguous 16B (async) copies and both
// WMMA operands load contiguously.
// ---------------------------------------------------------------------------
__global__ __launch_bounds__(256) void proj_kernel(
    const _Float16* __restrict__ A, const _Float16* __restrict__ WT16,
    const float* __restrict__ bo, float* __restrict__ out) {
  __shared__ __align__(16) _Float16 as[128][72];     // [row][k]
  __shared__ __align__(16) _Float16 bsT[64][72];     // [n][k]
  const int tid = threadIdx.x, wave = tid >> 5, lane = tid & 31;
  const int ct = blockIdx.x, rt = blockIdx.y;
  v8f acc[4] = {};

  for (int kb = 0; kb < EMBED/64; ++kb) {
#pragma unroll
    for (int i = 0; i < 4; ++i) {                    // 128x64 halves
      int j = tid + i*256;
      int row = j >> 3, c = (j & 7) << 3;
      copy16_g2l(A + (size_t)(rt*128 + row)*EMBED + kb*64 + c, &as[row][c]);
    }
#pragma unroll
    for (int i = 0; i < 2; ++i) {                    // 64x64 halves (N-major)
      int j = tid + i*256;
      int row = j >> 3, c = (j & 7) << 3;
      copy16_g2l(WT16 + (size_t)(ct*64 + row)*EMBED + kb*64 + c, &bsT[row][c]);
    }
    tile_join();
    v16h a0 = load_a_frag(&as[wave*16][0],  72);
    v16h a1 = load_a_frag(&as[wave*16][32], 72);
#pragma unroll
    for (int t = 0; t < 4; ++t) {
      acc[t] = wmma16(a0, load_bT_frag(&bsT[t*16][0],  72), acc[t]);
      acc[t] = wmma16(a1, load_bT_frag(&bsT[t*16][32], 72), acc[t]);
    }
    __syncthreads();
  }
#pragma unroll
  for (int t = 0; t < 4; ++t) {
    int col = ct*64 + t*16 + (lane & 15);
    float bias = bo[col];
#pragma unroll
    for (int r = 0; r < 8; ++r) {
      int mrow = r + ((lane & 16) ? 8 : 0);
      out[(size_t)(rt*128 + wave*16 + mrow)*EMBED + col] = acc[t][r] + bias;
    }
  }
}

extern "C" void kernel_launch(void* const* d_in, const int* in_sizes, int n_in,
                              void* d_out, int out_size, void* d_ws, size_t ws_size,
                              hipStream_t stream) {
  (void)in_sizes; (void)n_in; (void)out_size; (void)ws_size;
  const float* x  = (const float*)d_in[0];
  const float* Wq = (const float*)d_in[1];
  const float* Wk = (const float*)d_in[2];
  const float* Wv = (const float*)d_in[3];
  const float* Wo = (const float*)d_in[4];
  const float* bo = (const float*)d_in[5];
  float* out = (float*)d_out;

  const size_t NQ = (size_t)BATCH * HEADS * SEQ * HDIM;   // 4,194,304
  _Float16* Qb   = (_Float16*)d_ws;                       // [B,H,S,64]
  _Float16* Kb   = Qb + NQ;
  _Float16* Vb   = Kb + NQ;
  _Float16* Ab   = Vb + NQ;                               // [B*S, EMBED]
  _Float16* WoT16 = Ab + (size_t)BATCH * SEQ * EMBED;     // [EMBED][EMBED] (T)

  transpose_cvt<<<dim3(32, 32), 256, 0, stream>>>(Wo, WoT16);
  qkv_kernel<<<NROWS/128, 256, 0, stream>>>(x, Wq, Wk, Wv, Qb, Kb, Vb);
  attn_kernel<<<dim3(SEQ/128, BATCH*HEADS), 256, 0, stream>>>(Qb, Kb, Vb, Ab);
  proj_kernel<<<dim3(EMBED/64, (BATCH*SEQ)/128), 256, 0, stream>>>(Ab, WoT16, bo, out);
}
